// Head_79963701117489
// MI455X (gfx1250) — matrix operand order repaired and checked by reading, not compiled, not run on hardware
//
#include <hip/hip_runtime.h>
#include <hip/hip_bf16.h>

// ---- problem constants (from reference) ----
#define HEAD   64
#define CEMB   1024
#define BATCH  8
#define SEQ    4096

// ---- ext-vector types (trivial, union-safe) ----
typedef __attribute__((ext_vector_type(16))) __bf16       v16bf;
typedef __attribute__((ext_vector_type(8)))  float        v8f;
typedef __attribute__((ext_vector_type(4)))  unsigned int u32x4;
typedef __attribute__((ext_vector_type(8)))  int          i32x8;
typedef __attribute__((ext_vector_type(4)))  int          i32x4;
typedef __attribute__((ext_vector_type(4)))  float        f32x4;

union Frag16 {
    v16bf v;
    u32x4 q[2];
};

__device__ __forceinline__ float half16_max(float v) {
#pragma unroll
    for (int off = 8; off >= 1; off >>= 1)
        v = fmaxf(v, __shfl_xor(v, off, 16));
    return v;
}
__device__ __forceinline__ float half16_sum(float v) {
#pragma unroll
    for (int off = 8; off >= 1; off >>= 1)
        v += __shfl_xor(v, off, 16);
    return v;
}

// ---------------------------------------------------------------------
// TDM 2D tile load: global (bf16, row-major, row length = stride0 elems)
// -> LDS (rows compacted to tile_d0 elems). D# per CDNA5 ISA ch.8.
// 6-arg builtin form (this toolchain): (g0, g1, g2, g3, gx, cpol).
// ---------------------------------------------------------------------
__device__ __forceinline__ void tdm_load_2d(
    unsigned lds_addr, unsigned long long gaddr,
    unsigned tensor_d0, unsigned tensor_d1, unsigned long long stride0,
    unsigned tile_d0, unsigned tile_d1)
{
    u32x4 g0;
    g0.x = 0x1u;                                        // count=1, user D#
    g0.y = lds_addr;                                    // LDS byte address
    g0.z = (unsigned)(gaddr & 0xFFFFFFFFu);             // global_addr[31:0]
    g0.w = (unsigned)((gaddr >> 32) & 0x01FFFFFFu)      // global_addr[56:32]
         | (2u << 30);                                  // type = 2 ("image")
    i32x8 g1;
    g1[0] = (int)(1u << 16);                            // data_size = 2 bytes
    g1[1] = (int)((tensor_d0 & 0xFFFFu) << 16);         // tensor_dim0[15:0]
    g1[2] = (int)((tensor_d0 >> 16) | ((tensor_d1 & 0xFFFFu) << 16));
    g1[3] = (int)((tensor_d1 >> 16) | ((tile_d0 & 0xFFFFu) << 16));
    g1[4] = (int)(tile_d1 & 0xFFFFu);                   // tile_dim1, tile_dim2=0
    g1[5] = (int)(stride0 & 0xFFFFFFFFu);               // tensor_dim0_stride lo
    g1[6] = (int)((stride0 >> 32) & 0xFFFFu);           // stride hi, dim1_stride=0
    g1[7] = 0;
    const i32x4 z4 = {0, 0, 0, 0};                      // 2D: groups 2/3 unused
    const i32x8 z8 = {0, 0, 0, 0, 0, 0, 0, 0};
    __builtin_amdgcn_tensor_load_to_lds(g0, g1, z4, z4, z8, 0);
}

__device__ __forceinline__ unsigned lds_off(const void* p) {
    return (unsigned)(unsigned long long)(uintptr_t)p;  // addr[31:0] = LDS offset
}

// =====================================================================
// Kernel 0: one-time W fp32 -> bf16, transposed to [mat][n][k]
// =====================================================================
__global__ __launch_bounds__(256) void convert_w_kernel(
    const float* __restrict__ Wk, const float* __restrict__ Wq,
    const float* __restrict__ Wv, __bf16* __restrict__ Wt)
{
    const int i = blockIdx.x * 256 + threadIdx.x;       // 3*64*1024 total
    const int w = i >> 16;
    const int j = i & 65535;
    const int n = j >> 10;
    const int k = j & 1023;
    const float* wp = (w == 0) ? Wk : ((w == 1) ? Wq : Wv);
    Wt[(size_t)w * (HEAD * CEMB) + (size_t)n * CEMB + k] =
        (__bf16)wp[(size_t)k * HEAD + n];
}

// =====================================================================
// Kernel 1: fused QKV projection -> bf16 Q,K row-major; V transposed.
// Weight chunks staged by TDM; x staged via VALU (needs fp32->bf16).
// =====================================================================
__global__ __launch_bounds__(256) void proj_qkv_kernel(
    const float* __restrict__ x,
    const __bf16* __restrict__ Wt,       // [3][64][1024] bf16, n-major
    __bf16* __restrict__ Kb,
    __bf16* __restrict__ Qb,
    __bf16* __restrict__ Vt)             // [B][64][SEQ]
{
    __shared__ __align__(16) __bf16 xlds[128][32];      // [row][k]
    __shared__ __align__(16) __bf16 wlds[3][64][32];    // [mat][n][k]

    const int tid  = threadIdx.x;
    const int lane = tid & 31;
    const int wid  = tid >> 5;
    const int row0 = blockIdx.x * 128;
    const int m    = lane & 15;
    const int g    = lane >> 4;

    v8f acc[3][4] = {};

    for (int kc = 0; kc < 32; ++kc) {
        const int k0 = kc * 32;

        // ---- TDM: stage 3 weight chunks (64 n-rows x 32 k) ----
        if (wid == 0) {
#pragma unroll
            for (int w = 0; w < 3; ++w) {
                tdm_load_2d(lds_off(&wlds[w][0][0]),
                            (unsigned long long)(uintptr_t)(Wt + (size_t)w * (HEAD * CEMB) + k0),
                            CEMB, HEAD, CEMB, 32, 64);
            }
        }
        // ---- stage x chunk (128x32) fp32 -> bf16 ----
#pragma unroll
        for (int i = 0; i < 4; ++i) {
            const int idx4 = tid + i * 256;
            const int r  = idx4 >> 3;
            const int kk = (idx4 & 7) * 4;
            f32x4 f = *(const f32x4*)&x[(size_t)(row0 + r) * CEMB + k0 + kk];
            xlds[r][kk + 0] = (__bf16)f.x;
            xlds[r][kk + 1] = (__bf16)f.y;
            xlds[r][kk + 2] = (__bf16)f.z;
            xlds[r][kk + 3] = (__bf16)f.w;
        }
        if (wid == 0) __builtin_amdgcn_s_wait_tensorcnt(0);
        __syncthreads();

        // ---- A fragment: 16x32 bf16 ----
        Frag16 a;
        a.q[0] = *(const u32x4*)&xlds[wid * 16 + m][g * 8];
        a.q[1] = *(const u32x4*)&xlds[wid * 16 + m][16 + g * 8];

#pragma unroll
        for (int w = 0; w < 3; ++w) {
#pragma unroll
            for (int nt = 0; nt < 4; ++nt) {
                Frag16 bf;
                bf.q[0] = *(const u32x4*)&wlds[w][nt * 16 + m][g * 8];
                bf.q[1] = *(const u32x4*)&wlds[w][nt * 16 + m][16 + g * 8];
                acc[w][nt] = __builtin_amdgcn_wmma_f32_16x16x32_bf16(
                    false, a.v, false, bf.v, (short)0, acc[w][nt], false, false);
            }
        }
        __syncthreads();
    }

    // ---- stores: K,Q row-major; V transposed per batch ----
#pragma unroll
    for (int nt = 0; nt < 4; ++nt) {
#pragma unroll
        for (int r = 0; r < 8; ++r) {
            const int rowm = row0 + wid * 16 + r + 8 * g;
            const int n    = nt * 16 + m;
            Kb[(size_t)rowm * HEAD + n] = (__bf16)acc[0][nt][r];
            Qb[(size_t)rowm * HEAD + n] = (__bf16)(acc[1][nt][r] * 0.125f);
            const int bb = rowm >> 12;                 // rowm / SEQ
            const int tt = rowm & (SEQ - 1);
            Vt[((size_t)bb * HEAD + n) * SEQ + tt] = (__bf16)acc[2][nt][r];
        }
    }
}

// =====================================================================
// Kernel 2: causal flash attention; double-buffered TDM K/V staging
// overlapped with WMMA compute (TDM ops of a wave complete in order,
// so tensorcnt<=2 after issuing the next tile means current tile done).
// =====================================================================
__global__ __launch_bounds__(256) void flash_attn_kernel(
    const __bf16* __restrict__ Kb,
    const __bf16* __restrict__ Qb,
    const __bf16* __restrict__ Vt,       // [B][64][SEQ]
    float* __restrict__ out)
{
    __shared__ __align__(16) __bf16 klds[2][32][64];    // [buf][key][d]
    __shared__ __align__(16) __bf16 vldsT[2][64][32];   // [buf][d][key]
    __shared__ __align__(16) __bf16 plds[8][16][32];    // per-wave P reshape

    const int tid  = threadIdx.x;
    const int lane = tid & 31;
    const int wid  = tid >> 5;
    const int b    = blockIdx.x >> 5;
    const int tile = blockIdx.x & 31;
    const int q0   = tile * 128 + wid * 16;
    const int m    = lane & 15;
    const int g    = lane >> 4;

    const size_t bbase = (size_t)b * SEQ * HEAD;
    const __bf16* Kbb = Kb + bbase;
    const __bf16* Vtb = Vt + (size_t)b * HEAD * SEQ;

    // ---- Q A-fragments straight from global ----
    Frag16 qa[2];
    {
        const __bf16* qrow = Qb + bbase + (size_t)(q0 + m) * HEAD;
        qa[0].q[0] = *(const u32x4*)(qrow + g * 8);
        qa[0].q[1] = *(const u32x4*)(qrow + 16 + g * 8);
        qa[1].q[0] = *(const u32x4*)(qrow + 32 + g * 8);
        qa[1].q[1] = *(const u32x4*)(qrow + 48 + g * 8);
    }

    v8f oacc[4] = {};
    float mi[8], li[8];
#pragma unroll
    for (int r = 0; r < 8; ++r) { mi[r] = -3.0e38f; li[r] = 0.0f; }

    const int nkb  = tile * 4 + 4;
    const int mykb = (q0 + 15) >> 5;

    // ---- prologue: start DMA of tile 0 into buffer 0 ----
    if (wid == 0) {
        tdm_load_2d(lds_off(&klds[0][0][0]),
                    (unsigned long long)(uintptr_t)Kbb,
                    HEAD, SEQ, HEAD, 64, 32);
        tdm_load_2d(lds_off(&vldsT[0][0][0]),
                    (unsigned long long)(uintptr_t)Vtb,
                    SEQ, HEAD, SEQ, 32, 64);
    }

    for (int kb = 0; kb < nkb; ++kb) {
        const int kblk = kb * 32;
        const int buf  = kb & 1;

        if (wid == 0) {
            if (kb + 1 < nkb) {       // issue next tile, then wait current
                const int nblk = kblk + 32;
                tdm_load_2d(lds_off(&klds[buf ^ 1][0][0]),
                            (unsigned long long)(uintptr_t)(Kbb + (size_t)nblk * HEAD),
                            HEAD, SEQ, HEAD, 64, 32);
                tdm_load_2d(lds_off(&vldsT[buf ^ 1][0][0]),
                            (unsigned long long)(uintptr_t)(Vtb + nblk),
                            SEQ, HEAD, SEQ, 32, 64);
                __builtin_amdgcn_s_wait_tensorcnt(2);   // tile kb landed
            } else {
                __builtin_amdgcn_s_wait_tensorcnt(0);   // last tile landed
            }
        }
        if (kb + 2 < nkb) {           // global_prefetch_b8 two tiles ahead
            __builtin_prefetch(Kbb + (size_t)(kblk + 64) * HEAD, 0, 0);
        }
        __syncthreads();

        if (kb <= mykb) {             // wave-uniform: EXEC all-1s for WMMA
            // ---- S = Q K^T ----
            v8f s[2] = {};
#pragma unroll
            for (int t = 0; t < 2; ++t) {
#pragma unroll
                for (int dc = 0; dc < 2; ++dc) {
                    Frag16 kf;
                    kf.q[0] = *(const u32x4*)&klds[buf][t * 16 + m][dc * 32 + g * 8];
                    kf.q[1] = *(const u32x4*)&klds[buf][t * 16 + m][dc * 32 + 16 + g * 8];
                    s[t] = __builtin_amdgcn_wmma_f32_16x16x32_bf16(
                        false, qa[dc].v, false, kf.v, (short)0, s[t], false, false);
                }
            }

            // ---- causal mask + online softmax ----
#pragma unroll
            for (int r = 0; r < 8; ++r) {
                const int qidx = q0 + r + 8 * g;
                float s0 = ((kblk + m)      <= qidx) ? s[0][r] : -3.0e38f;
                float s1 = ((kblk + 16 + m) <= qidx) ? s[1][r] : -3.0e38f;
                const float rm    = half16_max(fmaxf(s0, s1));
                const float mnew  = fmaxf(mi[r], rm);
                const float alpha = __expf(mi[r] - mnew);
                const float p0 = __expf(s0 - mnew);
                const float p1 = __expf(s1 - mnew);
                const float rs = half16_sum(p0 + p1);
                li[r] = li[r] * alpha + rs;
                mi[r] = mnew;
#pragma unroll
                for (int nt = 0; nt < 4; ++nt) oacc[nt][r] *= alpha;
                plds[wid][r + 8 * g][m]      = (__bf16)p0;
                plds[wid][r + 8 * g][16 + m] = (__bf16)p1;
            }
            asm volatile("s_wait_dscnt 0" ::: "memory");

            // ---- O += P V ----
            Frag16 pa;
            pa.q[0] = *(const u32x4*)&plds[wid][m][g * 8];
            pa.q[1] = *(const u32x4*)&plds[wid][m][16 + g * 8];
#pragma unroll
            for (int nt = 0; nt < 4; ++nt) {
                Frag16 vf;
                vf.q[0] = *(const u32x4*)&vldsT[buf][nt * 16 + m][g * 8];
                vf.q[1] = *(const u32x4*)&vldsT[buf][nt * 16 + m][16 + g * 8];
                oacc[nt] = __builtin_amdgcn_wmma_f32_16x16x32_bf16(
                    false, pa.v, false, vf.v, (short)0, oacc[nt], false, false);
            }
        }
        __syncthreads();
    }

    // ---- epilogue: one reciprocal per row, then scaled stores ----
#pragma unroll
    for (int r = 0; r < 8; ++r) {
        const float inv = 1.0f / li[r];
        const int qr = q0 + r + 8 * g;
#pragma unroll
        for (int nt = 0; nt < 4; ++nt) {
            out[bbase + (size_t)qr * HEAD + nt * 16 + m] = oacc[nt][r] * inv;
        }
    }
}

// =====================================================================
// launch
// =====================================================================
extern "C" void kernel_launch(void* const* d_in, const int* in_sizes, int n_in,
                              void* d_out, int out_size, void* d_ws, size_t ws_size,
                              hipStream_t stream) {
    const float* x  = (const float*)d_in[0];
    const float* Wk = (const float*)d_in[1];
    const float* Wq = (const float*)d_in[2];
    const float* Wv = (const float*)d_in[3];
    float* out = (float*)d_out;

    const size_t per = (size_t)BATCH * SEQ * HEAD;      // 2M elems = 4 MiB bf16
    __bf16* Kb = (__bf16*)d_ws;
    __bf16* Qb = Kb + per;
    __bf16* Vt = Qb + per;
    __bf16* Wt = Vt + per;                              // [3][64][1024]

    convert_w_kernel<<<768, 256, 0, stream>>>(Wk, Wq, Wv, Wt);
    proj_qkv_kernel<<<256, 256, 0, stream>>>(x, Wt, Kb, Qb, Vt);
    flash_attn_kernel<<<256, 256, 0, stream>>>(Kb, Qb, Vt, out);
}
